// PrefillLinearAttnLayer_28621662060759
// MI455X (gfx1250) — matrix operand order repaired and checked by reading, not compiled, not run on hardware
//
#include <hip/hip_runtime.h>
#include <hip/hip_bf16.h>
#include <math.h>

// ---------------------------------------------------------------------------
// Problem constants (from the reference)
// ---------------------------------------------------------------------------
constexpr int S        = 2048;
constexpr int HID      = 2048;
constexpr int NH       = 16;
constexpr int DK       = 128;
constexpr int DV       = 128;
constexpr int CS       = 64;        // chunk size
constexpr int NC       = S / CS;    // 32 chunks
constexpr int KEY_DIM  = NH * DK;   // 2048
constexpr int VAL_DIM  = NH * DV;   // 2048
constexpr int CONV_DIM = 2 * KEY_DIM + VAL_DIM; // 6144
constexpr int KTAPS    = 4;

typedef __bf16 bf16;
typedef __attribute__((ext_vector_type(16))) __bf16 v16bf;
typedef __attribute__((ext_vector_type(4)))  __bf16 v4bf;
typedef __attribute__((ext_vector_type(8)))  float  v8f;
typedef __attribute__((ext_vector_type(4)))  float  v4f;

__device__ __forceinline__ bf16  f2bf(float x) { return (bf16)x; }
__device__ __forceinline__ float bf2f(bf16 x)  { return (float)x; }
__device__ __forceinline__ float silu(float x) { return x / (1.f + __expf(-x)); }

// ---------------------------------------------------------------------------
// Generic LDS->LDS WMMA matmul helper (bf16 in, f32 out), wave-cooperative.
// Effective math: C[M][N] (+)= A_eff[M][K] @ B_eff[K][N]
//   A storage: transA ? A[k*lda + m] : A[m*lda + k]
//   B storage: transB ? B[n*ldb + k] : B[k*ldb + n]
// Fragment layouts used here (wave32; contiguous k, same in A and B):
//   A 16x32 bf16: lane m = l15, elem j -> k = (lane>=16)*16 + j
//   B 32x16 bf16: lane n = l15, elem j -> k = (lane>=16)*16 + j
//   C/D 16x16 f32: vgpr r -> M = r + 8*(lane>=16), N = l15
// M, N multiples of 16; K multiple of 32. Uniform control flow per wave
// (EXEC all-ones requirement for WMMA).
// ---------------------------------------------------------------------------
__device__ __forceinline__ void lds_mm(
    const bf16* __restrict__ A, int lda, int transA,
    const bf16* __restrict__ B, int ldb, int transB,
    float* __restrict__ C, int ldc,
    int M, int N, int K, int accum)
{
  const int tid  = threadIdx.x;
  const int wid  = tid >> 5;
  const int lane = tid & 31;
  const int nw   = blockDim.x >> 5;
  const int l15  = lane & 15;
  const int half = lane >> 4;
  const int tn   = N >> 4;
  const int ntiles = (M >> 4) * tn;
  for (int t = wid; t < ntiles; t += nw) {
    const int tm0 = (t / tn) << 4;
    const int tn0 = (t % tn) << 4;
    v8f acc = {};
    for (int kk = 0; kk < K; kk += 32) {
      v16bf av, bv;
      const int m = tm0 + l15;
      const int n = tn0 + l15;
#pragma unroll
      for (int j = 0; j < 16; ++j) {
        const int ka = kk + (half << 4) + j;
        av[j] = transA ? A[ka * lda + m] : A[m * lda + ka];
      }
#pragma unroll
      for (int j = 0; j < 16; ++j) {
        const int kb = kk + (half << 4) + j;
        bv[j] = transB ? B[n * ldb + kb] : B[kb * ldb + n];
      }
      acc = __builtin_amdgcn_wmma_f32_16x16x32_bf16(false, av, false, bv,
                                                    (short)0, acc, false, false);
    }
#pragma unroll
    for (int r = 0; r < 8; ++r) {
      const int m = tm0 + r + (half << 3);
      const int n = tn0 + l15;
      if (accum) C[m * ldc + n] += acc[r];
      else       C[m * ldc + n]  = acc[r];
    }
  }
}

// ---------------------------------------------------------------------------
// Dense GEMM: C[M][N] = A[M][K] @ W[N][K]^T   (f32 in memory; bf16 WMMA math)
// Block = 256 threads (8 waves). Block tile 64(M) x 128(N), K-step 32,
// double-buffered LDS: tile k+1 global_load_b128 clauses issue before the
// tile-k WMMA stream; a single barrier per K-step separates producer/consumer.
// Staging loads all six float4s before converting so the loads form one
// clause with partial loadcnt waits; the compute block gathers all five LDS
// fragments before the four back-to-back WMMAs (partial dscnt waits).
// OOB rows are clamped (a W row only feeds its own output column, and the
// final stores are bounds-guarded).
// ---------------------------------------------------------------------------
constexpr int GBM = 64, GBN = 128, GBK = 32, GLD = GBK + 8; // 80B rows, 16B aligned

__global__ void __launch_bounds__(256)
gemm_wmma_bf16(const float* __restrict__ A, const float* __restrict__ W,
               float* __restrict__ C, int M, int N, int K)
{
  __shared__ bf16 sA[2][GBM][GLD];
  __shared__ bf16 sB[2][GBN][GLD];

  const int tid  = threadIdx.x;
  const int wid  = tid >> 5;
  const int lane = tid & 31;
  const int l15  = lane & 15;
  const int half = lane >> 4;
  const int m0   = blockIdx.y * GBM;
  const int n0   = blockIdx.x * GBN;
  const int wm   = (wid & 3) * 16;   // wave row base within block tile
  const int wn   = (wid >> 2) * 64;  // wave col base within block tile
  const int r    = tid >> 2;         // staging row (0..63), 4 threads/row
  const int cc   = (tid & 3) * 8;    // staging col chunk (8 floats)

  int gmA = m0 + r;      if (gmA >= M) gmA = M - 1;
  int gn0 = n0 + r;      if (gn0 >= N) gn0 = N - 1;
  int gn1 = n0 + r + 64; if (gn1 >= N) gn1 = N - 1;
  const float* pA  = A + (size_t)gmA * K + cc;
  const float* pW0 = W + (size_t)gn0 * K + cc;
  const float* pW1 = W + (size_t)gn1 * K + cc;

  auto stage = [&](int buf, int k0) {
    // issue all global loads first (one clause, partial loadcnt waits)
    const v4f a0 = *(const v4f*)(pA  + k0);
    const v4f a1 = *(const v4f*)(pA  + k0 + 4);
    const v4f b0 = *(const v4f*)(pW0 + k0);
    const v4f b1 = *(const v4f*)(pW0 + k0 + 4);
    const v4f b2 = *(const v4f*)(pW1 + k0);
    const v4f b3 = *(const v4f*)(pW1 + k0 + 4);
    *(v4bf*)&sA[buf][r][cc]          = __builtin_convertvector(a0, v4bf);
    *(v4bf*)&sA[buf][r][cc + 4]      = __builtin_convertvector(a1, v4bf);
    *(v4bf*)&sB[buf][r][cc]          = __builtin_convertvector(b0, v4bf);
    *(v4bf*)&sB[buf][r][cc + 4]      = __builtin_convertvector(b1, v4bf);
    *(v4bf*)&sB[buf][r + 64][cc]     = __builtin_convertvector(b2, v4bf);
    *(v4bf*)&sB[buf][r + 64][cc + 4] = __builtin_convertvector(b3, v4bf);
  };

  v8f acc[4] = {};
  stage(0, 0);

  for (int k0 = 0; k0 < K; k0 += GBK) {
    const int cur = (k0 >> 5) & 1;
    __syncthreads();               // staged tile visible; prev reads complete
    if (k0 + GBK < K) stage(cur ^ 1, k0 + GBK);

    // gather all LDS fragments up front, then 4 back-to-back WMMAs
    v16bf av, bv0, bv1, bv2, bv3;
    const int mrow = wm + l15;
    const int nrow = wn + l15;
#pragma unroll
    for (int j = 0; j < 16; ++j) {
      const int ka = (half << 4) + j;
      av[j] = sA[cur][mrow][ka];
    }
#pragma unroll
    for (int j = 0; j < 16; ++j) {
      const int kb = (half << 4) + j;
      bv0[j] = sB[cur][nrow][kb];
      bv1[j] = sB[cur][nrow + 16][kb];
      bv2[j] = sB[cur][nrow + 32][kb];
      bv3[j] = sB[cur][nrow + 48][kb];
    }
    acc[0] = __builtin_amdgcn_wmma_f32_16x16x32_bf16(false, av, false, bv0,
                                                     (short)0, acc[0], false, false);
    acc[1] = __builtin_amdgcn_wmma_f32_16x16x32_bf16(false, av, false, bv1,
                                                     (short)0, acc[1], false, false);
    acc[2] = __builtin_amdgcn_wmma_f32_16x16x32_bf16(false, av, false, bv2,
                                                     (short)0, acc[2], false, false);
    acc[3] = __builtin_amdgcn_wmma_f32_16x16x32_bf16(false, av, false, bv3,
                                                     (short)0, acc[3], false, false);
  }

#pragma unroll
  for (int t = 0; t < 4; ++t) {
    const int n = n0 + wn + t * 16 + l15;
#pragma unroll
    for (int rr = 0; rr < 8; ++rr) {
      const int m = m0 + wm + rr + (half << 3);
      if (m < M && n < N) C[(size_t)m * N + n] = acc[t][rr];
    }
  }
}

// ---------------------------------------------------------------------------
// Depthwise causal conv (K=4) + SiLU over mixed (S x CONV_DIM), 4 ch/thread
// ---------------------------------------------------------------------------
__global__ void __launch_bounds__(256)
conv_silu_kernel(const float* __restrict__ mixed, const float* __restrict__ w,
                 float* __restrict__ out)
{
  const int idx4 = blockIdx.x * 256 + threadIdx.x;      // over S*CONV_DIM/4
  if (idx4 >= S * (CONV_DIM / 4)) return;
  const int s  = idx4 / (CONV_DIM / 4);
  const int c4 = (idx4 % (CONV_DIM / 4)) * 4;

  v4f wv[4];
#pragma unroll
  for (int j = 0; j < 4; ++j) wv[j] = *(const v4f*)(w + (size_t)(c4 + j) * KTAPS);

  v4f acc = {0.f, 0.f, 0.f, 0.f};
#pragma unroll
  for (int t = 0; t < KTAPS; ++t) {
    const int si = s - (KTAPS - 1) + t;
    if (si >= 0) {
      const v4f xv = *(const v4f*)(mixed + (size_t)si * CONV_DIM + c4);
#pragma unroll
      for (int j = 0; j < 4; ++j) acc[j] += xv[j] * wv[j][t];
    }
  }
  v4f o;
#pragma unroll
  for (int j = 0; j < 4; ++j) o[j] = silu(acc[j]);
  *(v4f*)(out + (size_t)s * CONV_DIM + c4) = o;
}

// ---------------------------------------------------------------------------
// L2-normalize q and k rows (length 128) in place. One warp per row; each
// lane owns 4 contiguous elements (float4 load/store).
// ---------------------------------------------------------------------------
__global__ void __launch_bounds__(256)
l2norm_kernel(float* __restrict__ qkvc)
{
  const int warp = blockIdx.x * 8 + (threadIdx.x >> 5);
  const int lane = threadIdx.x & 31;
  if (warp >= 2 * S * NH) return;
  const int which = warp >= S * NH;          // 0 = q, 1 = k
  const int r = warp - which * S * NH;
  const int s = r >> 4, h = r & 15;
  float* row = qkvc + (size_t)s * CONV_DIM + which * KEY_DIM + h * DK + lane * 4;
  v4f v = *(const v4f*)row;
  float ss = v[0]*v[0] + v[1]*v[1] + v[2]*v[2] + v[3]*v[3];
#pragma unroll
  for (int m = 16; m >= 1; m >>= 1) ss += __shfl_xor(ss, m, 32);
  const float rs = rsqrtf(ss + 1e-6f);
  *(v4f*)row = v * rs;
}

// ---------------------------------------------------------------------------
// beta = sigmoid(b); g = -exp(A_log) * softplus(a + dt_bias)   (S x NH each)
// ---------------------------------------------------------------------------
__global__ void __launch_bounds__(256)
gates_kernel(const float* __restrict__ b, const float* __restrict__ a,
             const float* __restrict__ dt_bias, const float* __restrict__ A_log,
             float* __restrict__ beta, float* __restrict__ g)
{
  const int idx = blockIdx.x * 256 + threadIdx.x;
  if (idx >= S * NH) return;
  const int h = idx & (NH - 1);
  beta[idx] = 1.f / (1.f + __expf(-b[idx]));
  const float x = a[idx] + dt_bias[h];
  const float sp = (x > 20.f) ? x : log1pf(__expf(x));
  g[idx] = -__expf(A_log[h]) * sp;
}

// ---------------------------------------------------------------------------
// Chunked gated delta rule core. One block per head, 8 waves, sequential over
// 32 chunks. State (128x128 f32) in LDS; (I-L)^{-1} via exact log-depth
// doubling (L is strictly-lower nilpotent, index <= 64):
//   T = I+L; A = L; repeat 5x: A = A@A; T = T + A@T;
// All matmuls via v_wmma_f32_16x16x32_bf16 through lds_mm().
// ---------------------------------------------------------------------------
// Dynamic LDS arena offsets (bytes)
constexpr int O_STATE  = 0;                       // 128*128 f32  = 65536
constexpr int O_STATEB = O_STATE  + 65536;        // 128*128 bf16 = 32768
constexpr int O_QS     = O_STATEB + 32768;        // 64*128 bf16  = 16384
constexpr int O_KS     = O_QS     + 16384;
constexpr int O_KB     = O_KS     + 16384;
constexpr int O_VB     = O_KB     + 16384;
constexpr int O_F1     = O_VB     + 16384;        // 64*128 f32 = 32768
constexpr int O_F2     = O_F1     + 32768;
constexpr int O_LF     = O_F2     + 32768;        // 64*64 f32 = 16384
constexpr int O_TF     = O_LF     + 16384;
constexpr int O_A2F    = O_TF     + 16384;
constexpr int O_ABF    = O_A2F    + 16384;        // 64*64 bf16 = 8192
constexpr int O_TBF    = O_ABF    + 8192;
constexpr int O_AIBF   = O_TBF    + 8192;
constexpr int O_GC     = O_AIBF   + 8192;         // 64 f32
constexpr int O_EG     = O_GC     + 256;
constexpr int O_EINV   = O_EG     + 256;
constexpr int CORE_SMEM = O_EINV  + 256;          // 303872 B (< 320 KB WGP LDS)

__global__ void __launch_bounds__(256)
core_kernel(const float* __restrict__ qkvc, const float* __restrict__ gbuf,
            const float* __restrict__ betabuf, float* __restrict__ core)
{
  extern __shared__ char smem[];
  float* state  = (float*)(smem + O_STATE);
  bf16*  stateB = (bf16*) (smem + O_STATEB);
  bf16*  qs     = (bf16*) (smem + O_QS);
  bf16*  ks     = (bf16*) (smem + O_KS);
  bf16*  kb     = (bf16*) (smem + O_KB);
  bf16*  vb     = (bf16*) (smem + O_VB);
  float* f1     = (float*)(smem + O_F1);
  float* f2     = (float*)(smem + O_F2);
  float* Lf     = (float*)(smem + O_LF);
  float* Tf     = (float*)(smem + O_TF);
  float* A2f    = (float*)(smem + O_A2F);
  bf16*  Abf    = (bf16*) (smem + O_ABF);
  bf16*  Tbf    = (bf16*) (smem + O_TBF);
  bf16*  aibf   = (bf16*) (smem + O_AIBF);
  float* gc     = (float*)(smem + O_GC);
  float* eg     = (float*)(smem + O_EG);
  float* einv   = (float*)(smem + O_EINV);

  const int h   = blockIdx.x;
  const int tid = threadIdx.x;
  const float scale = rsqrtf((float)DK);

  for (int i = tid; i < DK * DV; i += 256) state[i] = 0.f;
  __syncthreads();

  for (int c = 0; c < NC; ++c) {
    const int s0 = c * CS;

    // --- per-chunk gate cumsum (parallel load, single-thread LDS scan) ---
    if (tid < CS) eg[tid] = gbuf[(s0 + tid) * NH + h];
    __syncthreads();
    if (tid == 0) {
      float a = 0.f;
      for (int i = 0; i < CS; ++i) { a += eg[i]; gc[i] = a; }
    }
    __syncthreads();
    const float glast = gc[CS - 1];
    if (tid < CS) { eg[tid] = __expf(gc[tid]); einv[tid] = __expf(glast - gc[tid]); }
    __syncthreads();

    // --- build bf16 operand tiles (float4 loads, packed bf16 LDS stores) ---
    for (int x = tid; x < CS * (DK / 4); x += 256) {
      const int i  = x >> 5;           // chunk row
      const int d4 = (x & 31) * 4;
      const int s  = s0 + i;
      const float* row = qkvc + (size_t)s * CONV_DIM;
      const float bt = betabuf[s * NH + h];
      const v4f qv = *(const v4f*)(row + h * DK + d4);
      const v4f kv = *(const v4f*)(row + KEY_DIM + h * DK + d4);
      const v4f vv = *(const v4f*)(row + 2 * KEY_DIM + h * DV + d4);
      *(v4bf*)&qs[i * DK + d4] = __builtin_convertvector(qv * scale, v4bf);
      *(v4bf*)&ks[i * DK + d4] = __builtin_convertvector(kv,         v4bf);
      *(v4bf*)&kb[i * DK + d4] = __builtin_convertvector(kv * bt,    v4bf);
      *(v4bf*)&vb[i * DK + d4] = __builtin_convertvector(vv * bt,    v4bf);
    }
    __syncthreads();

    // Lf = kb @ k^T ;  A2f = (q*scale) @ k^T    (both 64x64, K=128)
    lds_mm(kb, DK, 0, ks, DK, 1, Lf,  CS, CS, CS, DK, 0);
    lds_mm(qs, DK, 0, ks, DK, 1, A2f, CS, CS, CS, DK, 0);
    __syncthreads();

    // decay-mask L (strict lower, negated) and ai (tril incl diag); T = I + L
    for (int x = tid; x < CS * CS; x += 256) {
      const int i = x >> 6, j = x & 63;
      const float dm = __expf(fminf(gc[i] - gc[j], 10.f));
      const float Lv = (j < i) ? (-Lf[x] * dm) : 0.f;
      Lf[x]  = Lv;
      Abf[x] = f2bf(Lv);
      Tf[x]  = Lv + ((i == j) ? 1.f : 0.f);
      const float av = (j <= i) ? (A2f[x] * dm) : 0.f;
      aibf[x] = f2bf(av);
    }
    __syncthreads();

    // attn = (I-L)^{-1} by doubling: 5 iters of {A=A@A; T=T+A@T}
    for (int it = 0; it < 5; ++it) {
      lds_mm(Abf, CS, 0, Abf, CS, 0, A2f, CS, CS, CS, CS, 0);
      __syncthreads();
      for (int x = tid; x < CS * CS; x += 256) {
        Tbf[x] = f2bf(Tf[x]);
        Abf[x] = f2bf(A2f[x]);
      }
      __syncthreads();
      lds_mm(Abf, CS, 0, Tbf, CS, 0, Tf, CS, CS, CS, CS, 1);
      __syncthreads();
    }
    for (int x = tid; x < CS * CS; x += 256)
      Tbf[x] = f2bf(fminf(fmaxf(Tf[x], -1000.f), 1000.f));
    __syncthreads();

    // f1 = attn @ vb  ("value")
    lds_mm(Tbf, CS, 0, vb, DV, 0, f1, DV, CS, DV, CS, 0);
    __syncthreads();
    // kb <- kb * exp(g)
    for (int x = tid; x < CS * DK; x += 256)
      kb[x] = f2bf(bf2f(kb[x]) * eg[x >> 7]);
    __syncthreads();
    // f2 = attn @ (kb*exp(g))  ("kcd")
    lds_mm(Tbf, CS, 0, kb, DK, 0, f2, DK, CS, DK, CS, 0);
    __syncthreads();
    // kb <- bf16(kcd); stateB <- bf16(state)
    for (int x = tid; x < CS * DK; x += 256) kb[x] = f2bf(f2[x]);
    for (int x = tid; x < DK * DV; x += 256) stateB[x] = f2bf(state[x]);
    __syncthreads();
    // f2 = kcd @ state
    lds_mm(kb, DK, 0, stateB, DV, 0, f2, DV, CS, DV, DK, 0);
    __syncthreads();
    // v_new (into vb) = value - kcd@state ; qs <- q*scale*exp(g) ; ks <- k*exp(glast-g)
    for (int x = tid; x < CS * DV; x += 256) vb[x] = f2bf(f1[x] - f2[x]);
    for (int x = tid; x < CS * DK; x += 256) {
      const int i = x >> 7;
      qs[x] = f2bf(bf2f(qs[x]) * eg[i]);
      ks[x] = f2bf(bf2f(ks[x]) * einv[i]);
    }
    __syncthreads();
    // out = qe @ state + ai @ v_new
    lds_mm(qs, DK, 0, stateB, DV, 0, f1, DV, CS, DV, DK, 0);
    __syncthreads();
    lds_mm(aibf, CS, 0, vb, DV, 0, f1, DV, CS, DV, CS, 1);
    __syncthreads();
    // write core chunk (layout: [S][NH][DV]) and decay state
    for (int x = tid; x < CS * DV; x += 256) {
      const int i = x >> 7, d = x & 127;
      core[(size_t)(s0 + i) * VAL_DIM + h * DV + d] = f1[x];
    }
    const float egl = __expf(glast);
    for (int x = tid; x < DK * DV; x += 256) state[x] *= egl;
    __syncthreads();
    // state += (k*exp(glast-g))^T @ v_new   (128x128, K=64)
    lds_mm(ks, DK, 1, vb, DV, 0, state, DV, DK, DV, CS, 1);
    __syncthreads();
  }
}

// ---------------------------------------------------------------------------
// xn = core * rsqrt(mean(core^2)+eps) * norm_w * silu(z) ; one warp per row
// ---------------------------------------------------------------------------
__global__ void __launch_bounds__(256)
postnorm_kernel(const float* __restrict__ core, const float* __restrict__ z,
                const float* __restrict__ norm_w, float* __restrict__ xn)
{
  const int warp = blockIdx.x * 8 + (threadIdx.x >> 5);
  const int lane = threadIdx.x & 31;
  if (warp >= S * NH) return;
  const float* crow = core + (size_t)warp * DV + lane * 4;
  const float* zrow = z    + (size_t)warp * DV + lane * 4;
  float*       xrow = xn   + (size_t)warp * DV + lane * 4;
  const v4f cv = *(const v4f*)crow;
  float ss = cv[0]*cv[0] + cv[1]*cv[1] + cv[2]*cv[2] + cv[3]*cv[3];
#pragma unroll
  for (int m = 16; m >= 1; m >>= 1) ss += __shfl_xor(ss, m, 32);
  const float rs = rsqrtf(ss / (float)DV + 1e-6f);
  const v4f zv = *(const v4f*)zrow;
  const v4f nw = *(const v4f*)(norm_w + lane * 4);
  v4f o;
#pragma unroll
  for (int u = 0; u < 4; ++u) o[u] = cv[u] * rs * nw[u] * silu(zv[u]);
  *(v4f*)xrow = o;
}

// ---------------------------------------------------------------------------
// Host launcher
// ---------------------------------------------------------------------------
extern "C" void kernel_launch(void* const* d_in, const int* in_sizes, int n_in,
                              void* d_out, int out_size, void* d_ws, size_t ws_size,
                              hipStream_t stream)
{
  (void)in_sizes; (void)n_in; (void)out_size; (void)ws_size;
  const float* hidden  = (const float*)d_in[0];
  const float* qkv_w   = (const float*)d_in[1];
  const float* z_w     = (const float*)d_in[2];
  const float* b_w     = (const float*)d_in[3];
  const float* a_w     = (const float*)d_in[4];
  const float* o_w     = (const float*)d_in[5];
  const float* conv_w  = (const float*)d_in[6];
  const float* dt_bias = (const float*)d_in[7];
  const float* A_log   = (const float*)d_in[8];
  const float* norm_w  = (const float*)d_in[9];
  float* out = (float*)d_out;

  // workspace layout (bytes)
  char* ws = (char*)d_ws;
  float* mixed_pre = (float*)(ws);                             //  50,331,648
  float* qkvc      = (float*)(ws + 50331648);                  //  50,331,648
  float* zbuf      = (float*)(ws + 100663296);                 //  16,777,216
  float* bbuf      = (float*)(ws + 117440512);                 //     131,072
  float* abuf      = (float*)(ws + 117571584);                 //     131,072
  float* betab     = (float*)(ws + 117702656);                 //     131,072
  float* gbuf      = (float*)(ws + 117833728);                 //     131,072
  float* corebuf   = (float*)(ws + 117964800);                 //  16,777,216
  float* xnbuf     = (float*)(ws + 134742016);                 //  16,777,216

  // 1) projections (WMMA GEMMs): C[M][N] = A[M][K] @ W[N][K]^T
  gemm_wmma_bf16<<<dim3(CONV_DIM / GBN, S / GBM), 256, 0, stream>>>(
      hidden, qkv_w, mixed_pre, S, CONV_DIM, HID);
  gemm_wmma_bf16<<<dim3(VAL_DIM / GBN, S / GBM), 256, 0, stream>>>(
      hidden, z_w, zbuf, S, VAL_DIM, HID);
  gemm_wmma_bf16<<<dim3(1, S / GBM), 256, 0, stream>>>(
      hidden, b_w, bbuf, S, NH, HID);
  gemm_wmma_bf16<<<dim3(1, S / GBM), 256, 0, stream>>>(
      hidden, a_w, abuf, S, NH, HID);

  // 2) depthwise causal conv + SiLU
  conv_silu_kernel<<<(S * (CONV_DIM / 4)) / 256, 256, 0, stream>>>(mixed_pre, conv_w,
                                                                   qkvc);

  // 3) l2norm of q,k rows (in place)
  l2norm_kernel<<<(2 * S * NH) / 8, 256, 0, stream>>>(qkvc);

  // 4) gates
  gates_kernel<<<(S * NH + 255) / 256, 256, 0, stream>>>(bbuf, abuf, dt_bias, A_log,
                                                         betab, gbuf);

  // 5) chunked gated delta rule core (one block per head)
  core_kernel<<<NH, 256, CORE_SMEM, stream>>>(qkvc, gbuf, betab, corebuf);

  // 6) RMSNorm * norm_w * silu(z)
  postnorm_kernel<<<(S * NH) / 8, 256, 0, stream>>>(corebuf, zbuf, norm_w, xnbuf);

  // 7) output projection
  gemm_wmma_bf16<<<dim3(HID / GBN, S / GBM), 256, 0, stream>>>(
      xnbuf, o_w, out, S, HID, VAL_DIM);
}
